// Encoder_90142773608954
// MI455X (gfx1250) — matrix operand order repaired and checked by reading, not compiled, not run on hardware
//
#include <hip/hip_runtime.h>
#include <hip/hip_bf16.h>
#include <stdint.h>

// ---------------------------------------------------------------------------
// LSTM encoder: B=512, T=128, EMBED=300 (pad 320), UNITS=512, gates 4U=2048.
// Per step: z = x_t @ kernel + h @ rec_kernel + bias  via bf16 WMMA, f32 acc.
// Step GEMM uses 2x2 register blocking: one wave -> 32x32 output block,
// 4 independent accumulator chains, half the L2 fragment traffic per WMMA.
// ---------------------------------------------------------------------------

#define BATCH   512
#define TSTEPS  128
#define EMBED   300
#define EPAD    320           // EMBED padded to multiple of 32
#define UNITS   512
#define GATES   2048          // 4*UNITS
#define VOCABSZ 50000

typedef __attribute__((ext_vector_type(16))) __bf16       v16bf;
typedef __attribute__((ext_vector_type(8)))  float        v8f;
typedef __attribute__((ext_vector_type(8)))  unsigned int v8u;

static __device__ __forceinline__ unsigned short f2bf(float f) {
    unsigned int u = __float_as_uint(f);
    // round-to-nearest-even
    unsigned int r = (u + 0x7FFFu + ((u >> 16) & 1u)) >> 16;
    return (unsigned short)r;
}

static __device__ __forceinline__ float sigmoidf(float x) {
    return 1.0f / (1.0f + __expf(-x));
}

// Load a 16x32 bf16 A fragment (row-major source, row stride `ld` shorts).
// Lane holds row M=lane&15; VGPR j packs the K pair at
//   k = kb + (j&3)*2 + (j>>2)*16 + (lane>>4)*8     (ISA 16-bit A layout)
static __device__ __forceinline__ v16bf load_afrag(const unsigned short* rowbase,
                                                   int kb, int half) {
    v8u au;
#pragma unroll
    for (int j = 0; j < 8; ++j) {
        int ka = kb + ((j & 3) * 2) + ((j >> 2) * 16) + half * 8;
        au[j] = *(const unsigned int*)(rowbase + ka);
    }
    return __builtin_bit_cast(v16bf, au);
}

// Load a 32x16 bf16 B fragment from a column-major weight array
// (column = output n, contiguous in K). Lane holds column N=lane&15;
// VGPR j packs the pair at kb + (j&3)*2 + (j>>2)*16 + half*8 (same as A).
static __device__ __forceinline__ v16bf load_bfrag(const unsigned short* colbase,
                                                   int kb, int half) {
    v8u bu;
#pragma unroll
    for (int j = 0; j < 8; ++j) {
        int km = kb + ((j & 3) * 2) + ((j >> 2) * 16) + half * 8;
        bu[j] = *(const unsigned int*)(colbase + km);
    }
    return __builtin_bit_cast(v16bf, bu);
}

// ---------------------------------------------------------------------------
// Prep: weights -> bf16, transposed to column-major [N][Kpad] so that a
// B-fragment VGPR (two consecutive K values of one column) is one dword load.
// ---------------------------------------------------------------------------
__global__ void prep_weights(const float* __restrict__ kern,   // [EMBED][GATES]
                             const float* __restrict__ reck,   // [UNITS][GATES]
                             unsigned short* __restrict__ kT,  // [GATES][EPAD]
                             unsigned short* __restrict__ rT)  // [GATES][UNITS]
{
    const int nK = GATES * EPAD;
    const int nR = GATES * UNITS;
    int i = blockIdx.x * blockDim.x + threadIdx.x;
    if (i < nK) {
        int n = i / EPAD, k = i % EPAD;
        float v = (k < EMBED) ? kern[(size_t)k * GATES + n] : 0.0f;
        kT[i] = f2bf(v);
    } else if (i < nK + nR) {
        int j = i - nK;
        int n = j / UNITS, k = j % UNITS;
        rT[j] = f2bf(reck[(size_t)k * GATES + n]);
    }
}

// Gather embeddings for all timesteps: xall[t][b][EPAD] in bf16, zero padded.
__global__ void gather_embed(const int* __restrict__ seq,       // [B][T]
                             const float* __restrict__ emb,     // [VOCAB][EMBED]
                             unsigned short* __restrict__ xall) // [T][B][EPAD]
{
    int i = blockIdx.x * blockDim.x + threadIdx.x;
    const int total = TSTEPS * BATCH * EPAD;
    if (i >= total) return;
    int e = i % EPAD;
    int r = i / EPAD;            // r = t*B + b
    int t = r / BATCH;
    int b = r % BATCH;
    float v = 0.0f;
    if (e < EMBED) {
        int idx = seq[(size_t)b * TSTEPS + t];
        v = emb[(size_t)idx * EMBED + e];
    }
    xall[i] = f2bf(v);
}

__global__ void init_state(const float* __restrict__ sh, const float* __restrict__ sc,
                           float* __restrict__ h, float* __restrict__ c,
                           unsigned short* __restrict__ hbf)
{
    int i = blockIdx.x * blockDim.x + threadIdx.x;
    if (i >= BATCH * UNITS) return;
    float hv = sh[i];
    h[i]   = hv;
    c[i]   = sc[i];
    hbf[i] = f2bf(hv);
}

// ---------------------------------------------------------------------------
// Per-step GEMM, 2x2 register-blocked:
// z[b][n] = bias[n] + x_t[b][:] @ kernel + h[b][:] @ rec_kernel
// One wave computes a 32x32 block (4 16x16 tiles). Grid = (16, 64).
// ---------------------------------------------------------------------------
__global__ __launch_bounds__(32)
void step_gemm(const unsigned short* __restrict__ xall,  // [T][B][EPAD]
               const unsigned short* __restrict__ kT,    // [GATES][EPAD]
               const unsigned short* __restrict__ rT,    // [GATES][UNITS]
               const unsigned short* __restrict__ hbf,   // [B][UNITS]
               const float* __restrict__ bias,           // [GATES]
               float* __restrict__ z,                    // [B][GATES]
               int t)
{
    const int lane  = threadIdx.x;          // 0..31
    const int m     = lane & 15;
    const int half  = lane >> 4;
    const int brow0 = blockIdx.x * 32;      // first batch row of this block
    const int ncol0 = blockIdx.y * 32;      // first gate column of this block

    const int n0 = ncol0 + m;               // column for tile j=0
    const int n1 = ncol0 + 16 + m;          // column for tile j=1

    // 4 accumulator tiles, each initialized with bias (depends only on N)
    float bv0 = bias[n0];
    float bv1 = bias[n1];
    v8f acc00, acc01, acc10, acc11;
#pragma unroll
    for (int r = 0; r < 8; ++r) {
        acc00[r] = bv0; acc01[r] = bv1;
        acc10[r] = bv0; acc11[r] = bv1;
    }

    // ---- x_t @ kernel : K = EPAD (10 k-blocks) ----
    {
        const unsigned short* arow0 = xall + ((size_t)t * BATCH + brow0      + m) * EPAD;
        const unsigned short* arow1 = xall + ((size_t)t * BATCH + brow0 + 16 + m) * EPAD;
        const unsigned short* bcol0 = kT + (size_t)n0 * EPAD;
        const unsigned short* bcol1 = kT + (size_t)n1 * EPAD;
#pragma unroll
        for (int kb = 0; kb < EPAD; kb += 32) {
            v16bf a0 = load_afrag(arow0, kb, half);
            v16bf a1 = load_afrag(arow1, kb, half);
            v16bf b0 = load_bfrag(bcol0, kb, half);
            v16bf b1 = load_bfrag(bcol1, kb, half);
            acc00 = __builtin_amdgcn_wmma_f32_16x16x32_bf16(false, a0, false, b0, (short)0, acc00, false, false);
            acc01 = __builtin_amdgcn_wmma_f32_16x16x32_bf16(false, a0, false, b1, (short)0, acc01, false, false);
            acc10 = __builtin_amdgcn_wmma_f32_16x16x32_bf16(false, a1, false, b0, (short)0, acc10, false, false);
            acc11 = __builtin_amdgcn_wmma_f32_16x16x32_bf16(false, a1, false, b1, (short)0, acc11, false, false);
        }
    }

    // ---- h @ rec_kernel : K = UNITS (16 k-blocks) ----
    {
        const unsigned short* arow0 = hbf + (size_t)(brow0      + m) * UNITS;
        const unsigned short* arow1 = hbf + (size_t)(brow0 + 16 + m) * UNITS;
        const unsigned short* bcol0 = rT + (size_t)n0 * UNITS;
        const unsigned short* bcol1 = rT + (size_t)n1 * UNITS;
#pragma unroll
        for (int kb = 0; kb < UNITS; kb += 32) {
            v16bf a0 = load_afrag(arow0, kb, half);
            v16bf a1 = load_afrag(arow1, kb, half);
            v16bf b0 = load_bfrag(bcol0, kb, half);
            v16bf b1 = load_bfrag(bcol1, kb, half);
            acc00 = __builtin_amdgcn_wmma_f32_16x16x32_bf16(false, a0, false, b0, (short)0, acc00, false, false);
            acc01 = __builtin_amdgcn_wmma_f32_16x16x32_bf16(false, a0, false, b1, (short)0, acc01, false, false);
            acc10 = __builtin_amdgcn_wmma_f32_16x16x32_bf16(false, a1, false, b0, (short)0, acc10, false, false);
            acc11 = __builtin_amdgcn_wmma_f32_16x16x32_bf16(false, a1, false, b1, (short)0, acc11, false, false);
        }
    }

    // ---- store 4 C tiles: VGPR r -> row M = r + 8*half, col = n ----
#pragma unroll
    for (int r = 0; r < 8; ++r) {
        int row0 = brow0 + r + 8 * half;       // tiles (i=0, j=0/1)
        int row1 = brow0 + 16 + r + 8 * half;  // tiles (i=1, j=0/1)
        z[(size_t)row0 * GATES + n0] = acc00[r];
        z[(size_t)row0 * GATES + n1] = acc01[r];
        z[(size_t)row1 * GATES + n0] = acc10[r];
        z[(size_t)row1 * GATES + n1] = acc11[r];
    }
}

// ---------------------------------------------------------------------------
// Per-step pointwise: gates, cell/hidden update, padding mask, outputs.
// ---------------------------------------------------------------------------
__global__ void step_pointwise(const float* __restrict__ z,     // [B][GATES]
                               const int* __restrict__ seq,     // [B][T]
                               float* __restrict__ h,           // [B][UNITS]
                               float* __restrict__ c,           // [B][UNITS]
                               unsigned short* __restrict__ hbf,// [B][UNITS]
                               float* __restrict__ out,         // [B][T][UNITS]
                               int t)
{
    int i = blockIdx.x * blockDim.x + threadIdx.x;
    if (i >= BATCH * UNITS) return;
    int b = i >> 9;          // /512
    int u = i & 511;

    const float* zb = z + (size_t)b * GATES;
    float zi = zb[u];
    float zf = zb[u + UNITS];
    float zg = zb[u + 2 * UNITS];
    float zo = zb[u + 3 * UNITS];

    float ig = sigmoidf(zi);
    float fg = sigmoidf(zf);
    float og = sigmoidf(zo);
    float gg = tanhf(zg);

    float c_old = c[i];
    float h_old = h[i];
    float c_new = fg * c_old + ig * gg;
    float h_new = og * tanhf(c_new);

    bool mask = (seq[(size_t)b * TSTEPS + t] != 0);
    h_new = mask ? h_new : h_old;
    c_new = mask ? c_new : c_old;

    h[i]   = h_new;
    c[i]   = c_new;
    hbf[i] = f2bf(h_new);
    out[((size_t)b * TSTEPS + t) * UNITS + u] = h_new;
}

// ---------------------------------------------------------------------------
// Host side
// ---------------------------------------------------------------------------
extern "C" void kernel_launch(void* const* d_in, const int* in_sizes, int n_in,
                              void* d_out, int out_size, void* d_ws, size_t ws_size,
                              hipStream_t stream) {
    (void)in_sizes; (void)n_in; (void)out_size; (void)ws_size;

    const int*   seq   = (const int*)  d_in[0];   // [B][T]
    const float* sh    = (const float*)d_in[1];   // [B][UNITS]
    const float* sc    = (const float*)d_in[2];   // [B][UNITS]
    const float* emb   = (const float*)d_in[3];   // [VOCAB][EMBED]
    const float* kern  = (const float*)d_in[4];   // [EMBED][GATES]
    const float* reck  = (const float*)d_in[5];   // [UNITS][GATES]
    const float* bias  = (const float*)d_in[6];   // [GATES]

    float* outf = (float*)d_out;

    // workspace layout (all offsets 256B aligned)
    char* ws = (char*)d_ws;
    size_t off = 0;
    unsigned short* kT   = (unsigned short*)(ws + off); off += (size_t)GATES * EPAD  * 2; // 1.25 MB
    unsigned short* rT   = (unsigned short*)(ws + off); off += (size_t)GATES * UNITS * 2; // 2.00 MB
    unsigned short* xall = (unsigned short*)(ws + off); off += (size_t)TSTEPS * BATCH * EPAD * 2; // 40 MB
    float*          zbuf = (float*)         (ws + off); off += (size_t)BATCH * GATES * 4; // 4 MB
    float*          hbuf = (float*)         (ws + off); off += (size_t)BATCH * UNITS * 4; // 1 MB
    float*          cbuf = (float*)         (ws + off); off += (size_t)BATCH * UNITS * 4; // 1 MB
    unsigned short* hbf  = (unsigned short*)(ws + off); off += (size_t)BATCH * UNITS * 2; // 0.5 MB

    // --- prep ---
    {
        int total = GATES * EPAD + GATES * UNITS;
        prep_weights<<<(total + 255) / 256, 256, 0, stream>>>(kern, reck, kT, rT);
    }
    {
        int total = TSTEPS * BATCH * EPAD;
        gather_embed<<<(total + 255) / 256, 256, 0, stream>>>(seq, emb, xall);
    }
    {
        int total = BATCH * UNITS;
        init_state<<<(total + 255) / 256, 256, 0, stream>>>(sh, sc, hbuf, cbuf, hbf);
    }

    // --- recurrence: 128 sequential steps ---
    dim3 ggrid(BATCH / 32, GATES / 32);   // (16, 64) 32x32 blocks, one wave each
    int  ptotal  = BATCH * UNITS;
    int  pblocks = (ptotal + 255) / 256;
    for (int t = 0; t < TSTEPS; ++t) {
        step_gemm<<<ggrid, 32, 0, stream>>>(xall, kT, rT, hbf, bias, zbuf, t);
        step_pointwise<<<pblocks, 256, 0, stream>>>(zbuf, seq, hbuf, cbuf, hbf, outf, t);
    }

    // --- final states appended after enc_output ---
    size_t encN = (size_t)BATCH * TSTEPS * UNITS;
    hipMemcpyAsync(outf + encN, hbuf, (size_t)BATCH * UNITS * sizeof(float),
                   hipMemcpyDeviceToDevice, stream);
    hipMemcpyAsync(outf + encN + (size_t)BATCH * UNITS, cbuf,
                   (size_t)BATCH * UNITS * sizeof(float),
                   hipMemcpyDeviceToDevice, stream);
}
